// MultiModalCNN_90039694394091
// MI455X (gfx1250) — 8-bit matrix operand order repaired and checked by reading, compiled, not run on hardware
//
#include <hip/hip_runtime.h>
#include <cstdint>


typedef __attribute__((ext_vector_type(8))) int v8i;

#define EPS_ABSMAX 1e-8f
#define BN_EPS     1e-5f

enum {
  AM_X = 0, AM_W0 = 1, AM_W1 = 2, AM_W2 = 3, AM_W3 = 4,
  AM_Z0 = 5, AM_Z1 = 6, AM_Z2 = 7, AM_Z3 = 8,
  SUM0 = 32,  SS0 = 64,
  SUM1 = 96,  SS1 = 160,
  SUM2 = 224, SS2 = 288,
  SUM3 = 352, SS3 = 356,
  ZERO_FLOATS = 384,
  A0 = 384, SH0 = 416,
  A1 = 448, SH1 = 512,
  A2 = 576, SH2 = 640,
  A3 = 704, SH3 = 708
};

static const size_t APACK0 = 64 * 1024;
static const size_t APACK1 = APACK0 + 8 * 1024;
static const size_t APACK2 = APACK1 + 16 * 1024;
static const size_t QW3OFF = APACK2 + 24 * 1024;
static const size_t XQ_A   = 128 * 1024;
static const size_t XQ_B   = XQ_A + (size_t)(18u << 20);
static const size_t YBUFO  = XQ_B + (size_t)(18u << 20);

__global__ void k_zero(float* p, int n) {
  int i = blockIdx.x * blockDim.x + threadIdx.x;
  if (i < n) p[i] = 0.f;
}

__global__ void k_absmax(const float* __restrict__ x, int n, unsigned* am) {
  int i = blockIdx.x * blockDim.x + threadIdx.x;
  int str = gridDim.x * blockDim.x;
  float m = 0.f;
  for (; i < n; i += str) m = fmaxf(m, fabsf(x[i]));
#pragma unroll
  for (int msk = 1; msk <= 16; msk <<= 1) m = fmaxf(m, __shfl_xor(m, msk, 32));
  if ((threadIdx.x & 31) == 0) atomicMax(am, __float_as_uint(m));
}

__global__ void k_quant(const float* __restrict__ x, int n,
                        const unsigned* am, int8_t* __restrict__ q) {
  int i = blockIdx.x * blockDim.x + threadIdx.x;
  if (i >= n) return;
  float s = fmaxf(__uint_as_float(*am), EPS_ABSMAX) * (1.f / 127.f);
  float qf = rintf(x[i] / s);
  qf = fminf(fmaxf(qf, -127.f), 127.f);
  q[i] = (int8_t)(int)qf;
}

__global__ void k_pack_w(const float* __restrict__ w, const unsigned* am,
                         unsigned* __restrict__ apack,
                         int KC, int CIN, int KW, int ndw) {
  int f = blockIdx.x * blockDim.x + threadIdx.x;
  if (f >= ndw) return;
  int v = f & 7;
  int lane = (f >> 3) & 31;
  int kcmt = f >> 8;
  int kc = kcmt % KC;
  int mt = kcmt / KC;
  int m = lane & 15, h = lane >> 4;
  float sw = fmaxf(__uint_as_float(*am), EPS_ABSMAX) * (1.f / 127.f);
  int off = 32 * (v >> 2) + 16 * ((v >> 1) & 1) + 4 * (v & 1) + 8 * h;
  unsigned dw = 0;
#pragma unroll
  for (int j = 0; j < 4; ++j) {
    int kk = kc * 64 + off + j;
    int q = 0;
    if (kk < CIN * KW) {
      int ci = kk / KW, k = kk - ci * KW;
      int co = mt * 16 + m;
      float qf = rintf(w[(co * CIN + ci) * KW + k] / sw);
      qf = fminf(fmaxf(qf, -127.f), 127.f);
      q = (int)qf;
    }
    dw |= ((unsigned)(q & 0xFF)) << (8 * j);
  }
  apack[f] = dw;
}

template <int CIN, int KW, int COUT, int KC>
__global__ void __launch_bounds__(32)
k_conv_wmma(const int8_t* __restrict__ xq, const unsigned* amx,
            const unsigned* amw, const int8_t* __restrict__ Apack,
            const float* __restrict__ bias, float* __restrict__ yout,
            float* __restrict__ sum, float* __restrict__ sumsq, int L) {
  constexpr int CMT = COUT / 16;
  constexpr int NT = 4;
  constexpr int PAD = (KW - 1) / 2;
  constexpr int TW = 64 + KW - 1;
  constexpr int TWP = 72;

  __shared__ int8_t tile[CIN * TWP];

  const int segs = L >> 6;
  const int b = blockIdx.x / segs;
  const int l0 = (blockIdx.x % segs) * 64;
  const int lane = threadIdx.x & 31;
  const int n = lane & 15, h = lane >> 4;

  for (int i = threadIdx.x; i < CIN * TW; i += 32) {
    int ci = i / TW, li = i - ci * TW;
    int l = l0 - PAD + li;
    int8_t v = 0;
    if (l >= 0 && l < L) v = xq[((size_t)b * CIN + ci) * L + l];
    tile[ci * TWP + li] = v;
  }
  __syncthreads();

  v8i acc[CMT][NT];
#pragma unroll
  for (int mt = 0; mt < CMT; ++mt)
#pragma unroll
    for (int nt = 0; nt < NT; ++nt)
#pragma unroll
      for (int r = 0; r < 8; ++r) acc[mt][nt][r] = 0;

#pragma unroll
  for (int kc = 0; kc < KC; ++kc) {
    if (kc + 1 < KC)
      __builtin_prefetch(Apack + ((size_t)((kc + 1) * 32 + lane)) * 32, 0, 1);

    v8i bm[NT];
#pragma unroll
    for (int v = 0; v < 8; ++v) {
      const int kb = 32 * (v >> 2) + 16 * ((v >> 1) & 1) + 4 * (v & 1) + 8 * h;
      unsigned dw0 = 0, dw1 = 0, dw2 = 0, dw3 = 0;
#pragma unroll
      for (int j = 0; j < 4; ++j) {
        int kk = kc * 64 + kb + j;
        if (kk < CIN * KW) {
          int ci = kk / KW, k = kk - ci * KW;
          int base = ci * TWP + k + n;
          dw0 |= ((unsigned)(uint8_t)tile[base +  0]) << (8 * j);
          dw1 |= ((unsigned)(uint8_t)tile[base + 16]) << (8 * j);
          dw2 |= ((unsigned)(uint8_t)tile[base + 32]) << (8 * j);
          dw3 |= ((unsigned)(uint8_t)tile[base + 48]) << (8 * j);
        }
      }
      bm[0][v] = (int)dw0; bm[1][v] = (int)dw1;
      bm[2][v] = (int)dw2; bm[3][v] = (int)dw3;
    }

#pragma unroll
    for (int mt = 0; mt < CMT; ++mt) {
      v8i a = *(const v8i*)(Apack + ((size_t)((mt * KC + kc) * 32 + lane)) * 32);
#pragma unroll
      for (int nt = 0; nt < NT; ++nt)
        acc[mt][nt] = __builtin_amdgcn_wmma_i32_16x16x64_iu8(
            true, a, true, bm[nt], acc[mt][nt], false, false);
    }
  }

  const float sx = fmaxf(__uint_as_float(*amx), EPS_ABSMAX) * (1.f / 127.f);
  const float sw = fmaxf(__uint_as_float(*amw), EPS_ABSMAX) * (1.f / 127.f);
  const float sf = sx * sw;

#pragma unroll
  for (int mt = 0; mt < CMT; ++mt)
#pragma unroll
    for (int r = 0; r < 8; ++r) {
      const int co = mt * 16 + r + 8 * h;
      const float bv = bias[co];
      float s = 0.f, s2 = 0.f;
#pragma unroll
      for (int nt = 0; nt < NT; ++nt) {
        float yv = (float)acc[mt][nt][r] * sf + bv;
        yout[((size_t)b * COUT + co) * L + (l0 + nt * 16 + n)] = yv;
        s += yv;
        s2 += yv * yv;
      }
#pragma unroll
      for (int msk = 1; msk <= 8; msk <<= 1) {
        s += __shfl_xor(s, msk, 32);
        s2 += __shfl_xor(s2, msk, 32);
      }
      if (n == 0) {
        atomicAdd(&sum[co], s);
        atomicAdd(&sumsq[co], s2);
      }
    }
}

__global__ void k_bn_fin(const float* sum, const float* ss, const float* g,
                         const float* be, int C, float invcnt,
                         float* aout, float* shout) {
  int c = threadIdx.x;
  if (c < C) {
    float mean = sum[c] * invcnt;
    float var = fmaxf(ss[c] * invcnt - mean * mean, 0.f);
    float a = g[c] * rsqrtf(var + BN_EPS);
    aout[c] = a;
    shout[c] = be[c] - mean * a;
  }
}

__global__ void k_bn_relu_absmax(const float* __restrict__ y, const float* a,
                                 const float* sh, int total, int C, int L,
                                 unsigned* am) {
  int i = blockIdx.x * blockDim.x + threadIdx.x;
  int str = gridDim.x * blockDim.x;
  float m = 0.f;
  for (; i < total; i += str) {
    int c = (i / L) % C;
    float z = fmaxf(fmaf(y[i], a[c], sh[c]), 0.f);
    m = fmaxf(m, z);
  }
#pragma unroll
  for (int msk = 1; msk <= 16; msk <<= 1) m = fmaxf(m, __shfl_xor(m, msk, 32));
  if ((threadIdx.x & 31) == 0) atomicMax(am, __float_as_uint(m));
}

__global__ void k_quant_pool(const float* __restrict__ y, const float* a,
                             const float* sh, const unsigned* am,
                             int8_t* __restrict__ qout, int C, int Lin,
                             int total) {
  int i = blockIdx.x * blockDim.x + threadIdx.x;
  if (i >= total) return;
  int Lout = Lin >> 1;
  int lo = i % Lout;
  int bc = i / Lout;
  int c = bc % C;
  const float* yb = y + (size_t)bc * Lin + 2 * lo;
  float z0 = fmaxf(fmaf(yb[0], a[c], sh[c]), 0.f);
  float z1 = fmaxf(fmaf(yb[1], a[c], sh[c]), 0.f);
  float zp = fmaxf(z0, z1);
  float s = fmaxf(__uint_as_float(*am), EPS_ABSMAX) * (1.f / 127.f);
  float q = rintf(zp / s);
  q = fminf(fmaxf(q, -127.f), 127.f);
  qout[i] = (int8_t)(int)q;
}

__global__ void k_conv3(const int8_t* __restrict__ xq,
                        const int8_t* __restrict__ qw, const unsigned* amx,
                        const unsigned* amw, const float* __restrict__ bias,
                        float* __restrict__ yout, float* sum, float* sumsq) {
  int idx = blockIdx.x * blockDim.x + threadIdx.x;
  int b = idx >> 9, l = idx & 511;
  const int8_t* xb = xq + (size_t)b * 64 * 512 + l;
  int s0 = 0, s1 = 0, s2 = 0;
#pragma unroll
  for (int ci = 0; ci < 64; ++ci) {
    int xv = (int)xb[(size_t)ci * 512];
    s0 += xv * (int)qw[ci];
    s1 += xv * (int)qw[64 + ci];
    s2 += xv * (int)qw[128 + ci];
  }
  float sx = fmaxf(__uint_as_float(*amx), EPS_ABSMAX) * (1.f / 127.f);
  float sw = fmaxf(__uint_as_float(*amw), EPS_ABSMAX) * (1.f / 127.f);
  float sf = sx * sw;
  float y0 = (float)s0 * sf + bias[0];
  float y1 = (float)s1 * sf + bias[1];
  float y2 = (float)s2 * sf + bias[2];
  yout[((size_t)b * 3 + 0) * 512 + l] = y0;
  yout[((size_t)b * 3 + 1) * 512 + l] = y1;
  yout[((size_t)b * 3 + 2) * 512 + l] = y2;
  float p0 = y0, p1 = y1, p2 = y2, q0 = y0 * y0, q1 = y1 * y1, q2 = y2 * y2;
#pragma unroll
  for (int msk = 1; msk <= 16; msk <<= 1) {
    p0 += __shfl_xor(p0, msk, 32); q0 += __shfl_xor(q0, msk, 32);
    p1 += __shfl_xor(p1, msk, 32); q1 += __shfl_xor(q1, msk, 32);
    p2 += __shfl_xor(p2, msk, 32); q2 += __shfl_xor(q2, msk, 32);
  }
  if ((threadIdx.x & 31) == 0) {
    atomicAdd(&sum[0], p0); atomicAdd(&sumsq[0], q0);
    atomicAdd(&sum[1], p1); atomicAdd(&sumsq[1], q1);
    atomicAdd(&sum[2], p2); atomicAdd(&sumsq[2], q2);
  }
}

__global__ void k_final_mean(const float* __restrict__ y, const float* a,
                             const float* sh, const unsigned* am,
                             float* __restrict__ out) {
  __shared__ float red[128];
  int bo = blockIdx.x;
  int o = bo % 3;
  const float* yb = y + (size_t)bo * 512;
  float av = a[o], sv = sh[o];
  float s = fmaxf(__uint_as_float(*am), EPS_ABSMAX) * (1.f / 127.f);
  float acc = 0.f;
  for (int l = threadIdx.x; l < 512; l += 128) {
    float z = fmaxf(fmaf(yb[l], av, sv), 0.f);
    float q = fminf(fmaxf(rintf(z / s), -127.f), 127.f);
    acc += q * s;
  }
  red[threadIdx.x] = acc;
  __syncthreads();
  for (int w = 64; w >= 1; w >>= 1) {
    if ((int)threadIdx.x < w) red[threadIdx.x] += red[threadIdx.x + w];
    __syncthreads();
  }
  if (threadIdx.x == 0) out[bo] = red[0] * (1.f / 512.f);
}

extern "C" void kernel_launch(void* const* d_in, const int* in_sizes, int n_in,
                              void* d_out, int out_size, void* d_ws,
                              size_t ws_size, hipStream_t stream) {
  (void)in_sizes; (void)n_in; (void)out_size; (void)ws_size;
  const float* x   = (const float*)d_in[0];
  const float* w0  = (const float*)d_in[1];
  const float* b0  = (const float*)d_in[2];
  const float* g0  = (const float*)d_in[3];
  const float* be0 = (const float*)d_in[4];
  const float* w1  = (const float*)d_in[5];
  const float* b1  = (const float*)d_in[6];
  const float* g1  = (const float*)d_in[7];
  const float* be1 = (const float*)d_in[8];
  const float* w2  = (const float*)d_in[9];
  const float* b2  = (const float*)d_in[10];
  const float* g2  = (const float*)d_in[11];
  const float* be2 = (const float*)d_in[12];
  const float* w3  = (const float*)d_in[13];
  const float* b3  = (const float*)d_in[14];
  const float* g3  = (const float*)d_in[15];
  const float* be3 = (const float*)d_in[16];
  float* out = (float*)d_out;

  char* ws = (char*)d_ws;
  float* st = (float*)ws;
  unsigned* stu = (unsigned*)ws;
  unsigned* ap0 = (unsigned*)(ws + APACK0);
  unsigned* ap1 = (unsigned*)(ws + APACK1);
  unsigned* ap2 = (unsigned*)(ws + APACK2);
  int8_t* qw3 = (int8_t*)(ws + QW3OFF);
  int8_t* xq0 = (int8_t*)(ws + XQ_A);
  int8_t* xq1 = (int8_t*)(ws + XQ_B);
  int8_t* xq2 = (int8_t*)(ws + XQ_A);
  int8_t* xq3 = (int8_t*)(ws + XQ_B);
  float* ybuf = (float*)(ws + YBUFO);

  const int B = 256;

  k_zero<<<2, 256, 0, stream>>>(st, ZERO_FLOATS);

  const int nX = B * 5 * 4096;
  k_absmax<<<512, 256, 0, stream>>>(x, nX, stu + AM_X);
  k_quant<<<(nX + 255) / 256, 256, 0, stream>>>(x, nX, stu + AM_X, xq0);
  k_absmax<<<1, 256, 0, stream>>>(w0, 32 * 5 * 5, stu + AM_W0);
  k_absmax<<<4, 256, 0, stream>>>(w1, 64 * 32 * 5, stu + AM_W1);
  k_absmax<<<8, 256, 0, stream>>>(w2, 64 * 64 * 5, stu + AM_W2);
  k_absmax<<<1, 256, 0, stream>>>(w3, 3 * 64, stu + AM_W3);
  k_pack_w<<<2, 256, 0, stream>>>(w0, stu + AM_W0, ap0, 1, 5, 5, 2 * 1 * 256);
  k_pack_w<<<12, 256, 0, stream>>>(w1, stu + AM_W1, ap1, 3, 32, 5, 4 * 3 * 256);
  k_pack_w<<<20, 256, 0, stream>>>(w2, stu + AM_W2, ap2, 5, 64, 5, 4 * 5 * 256);
  k_quant<<<1, 256, 0, stream>>>(w3, 192, stu + AM_W3, qw3);

  k_conv_wmma<5, 5, 32, 1><<<B * (4096 / 64), 32, 0, stream>>>(
      xq0, stu + AM_X, stu + AM_W0, (const int8_t*)ap0, b0, ybuf,
      st + SUM0, st + SS0, 4096);
  k_bn_fin<<<1, 64, 0, stream>>>(st + SUM0, st + SS0, g0, be0, 32,
                                 1.f / (256.f * 4096.f), st + A0, st + SH0);
  k_bn_relu_absmax<<<1024, 256, 0, stream>>>(ybuf, st + A0, st + SH0,
                                             B * 32 * 4096, 32, 4096,
                                             stu + AM_Z0);
  k_quant_pool<<<(B * 32 * 2048 + 255) / 256, 256, 0, stream>>>(
      ybuf, st + A0, st + SH0, stu + AM_Z0, xq1, 32, 4096, B * 32 * 2048);

  k_conv_wmma<32, 5, 64, 3><<<B * (2048 / 64), 32, 0, stream>>>(
      xq1, stu + AM_Z0, stu + AM_W1, (const int8_t*)ap1, b1, ybuf,
      st + SUM1, st + SS1, 2048);
  k_bn_fin<<<1, 64, 0, stream>>>(st + SUM1, st + SS1, g1, be1, 64,
                                 1.f / (256.f * 2048.f), st + A1, st + SH1);
  k_bn_relu_absmax<<<1024, 256, 0, stream>>>(ybuf, st + A1, st + SH1,
                                             B * 64 * 2048, 64, 2048,
                                             stu + AM_Z1);
  k_quant_pool<<<(B * 64 * 1024 + 255) / 256, 256, 0, stream>>>(
      ybuf, st + A1, st + SH1, stu + AM_Z1, xq2, 64, 2048, B * 64 * 1024);

  k_conv_wmma<64, 5, 64, 5><<<B * (1024 / 64), 32, 0, stream>>>(
      xq2, stu + AM_Z1, stu + AM_W2, (const int8_t*)ap2, b2, ybuf,
      st + SUM2, st + SS2, 1024);
  k_bn_fin<<<1, 64, 0, stream>>>(st + SUM2, st + SS2, g2, be2, 64,
                                 1.f / (256.f * 1024.f), st + A2, st + SH2);
  k_bn_relu_absmax<<<1024, 256, 0, stream>>>(ybuf, st + A2, st + SH2,
                                             B * 64 * 1024, 64, 1024,
                                             stu + AM_Z2);
  k_quant_pool<<<(B * 64 * 512 + 255) / 256, 256, 0, stream>>>(
      ybuf, st + A2, st + SH2, stu + AM_Z2, xq3, 64, 1024, B * 64 * 512);

  k_conv3<<<B * 512 / 256, 256, 0, stream>>>(xq3, qw3, stu + AM_Z2,
                                             stu + AM_W3, b3, ybuf,
                                             st + SUM3, st + SS3);
  k_bn_fin<<<1, 64, 0, stream>>>(st + SUM3, st + SS3, g3, be3, 3,
                                 1.f / (256.f * 512.f), st + A3, st + SH3);
  k_bn_relu_absmax<<<256, 256, 0, stream>>>(ybuf, st + A3, st + SH3,
                                            B * 3 * 512, 3, 512, stu + AM_Z3);
  k_final_mean<<<B * 3, 128, 0, stream>>>(ybuf, st + A3, st + SH3,
                                          stu + AM_Z3, out);
}
